// MultiHeadAttention_19164144074875
// MI455X (gfx1250) — matrix operand order repaired and checked by reading, not compiled, not run on hardware
//
#include <hip/hip_runtime.h>

// ---------------------------------------------------------------------------
// MI455X (gfx1250) multi-head attention forward, bf16 WMMA everywhere.
// Shapes fixed by the reference: B=2, S=2048, D=2048, H=16, dk=128.
// ---------------------------------------------------------------------------
#define B_  2
#define S_  2048
#define D_  2048
#define H_  16
#define DK  128
#define M_  (B_ * S_)   // 4096 rows for the projections

typedef __attribute__((ext_vector_type(16))) __bf16 v16bf;
typedef __attribute__((ext_vector_type(8)))  float  v8f;

union BFrag16 {
  uint4          u[2];
  v16bf          v;
  unsigned short h[16];
};

__device__ __forceinline__ unsigned short f2bf(float f) {
  union { float f; unsigned int u; } x; x.f = f;
  unsigned int r = x.u + 0x7FFFu + ((x.u >> 16) & 1u);   // round-to-nearest-even
  return (unsigned short)(r >> 16);
}

__device__ __forceinline__ v8f wmma_bf16(v16bf a, v16bf b, v8f c) {
  // D(16x16,f32) = A(16x32,bf16) * B(32x16,bf16) + C
  return __builtin_amdgcn_wmma_f32_16x16x32_bf16(false, a, false, b, (short)0, c,
                                                 false, false);
}

// A fragment (16 rows x 32 K, bf16). ISA layout: lane m = l&15;
// elems 0..7  -> K = k0 + 8*hi + e ; elems 8..15 -> K = k0 + 16 + 8*hi + (e-8).
__device__ __forceinline__ v16bf load_fragA(const unsigned short* base, int stride,
                                            int row0, int k0) {
  const int lane = threadIdx.x & 31;
  const int hi   = lane >> 4;
  const unsigned short* p = base + (size_t)(row0 + (lane & 15)) * stride + k0 + hi * 8;
  BFrag16 f;
  f.u[0] = *(const uint4*)(p);
  f.u[1] = *(const uint4*)(p + 16);
  return f.v;
}

// B fragment (32 K x 16 cols, bf16). Lane col = l&15 (selects source row, since the
// source is row-major over K); elems 0..7 -> K = k0 + 8*hi + e ; elems 8..15 -> K = k0 + 16 + 8*hi + (e-8).
__device__ __forceinline__ v16bf load_fragB(const unsigned short* base, int stride,
                                            int col0, int k0) {
  const int lane = threadIdx.x & 31;
  const int hi   = lane >> 4;
  const unsigned short* p = base + (size_t)(col0 + (lane & 15)) * stride + k0 + hi * 8;
  BFrag16 f;
  f.u[0] = *(const uint4*)(p);
  f.u[1] = *(const uint4*)(p + 16);
  return f.v;
}

// ---------------------------------------------------------------------------
// fp32 -> bf16 bulk convert (4 elems / thread, counts are multiples of 1024)
// ---------------------------------------------------------------------------
__global__ __launch_bounds__(256)
void cvt_f32_bf16(const float* __restrict__ in, unsigned short* __restrict__ out, long n4) {
  long i = (long)blockIdx.x * blockDim.x + threadIdx.x;
  if (i < n4) {
    float4 f = ((const float4*)in)[i];
    ushort4 o;
    o.x = f2bf(f.x); o.y = f2bf(f.y); o.z = f2bf(f.z); o.w = f2bf(f.w);
    ((ushort4*)out)[i] = o;
  }
}

// ---------------------------------------------------------------------------
// Fused QKV projection + RoPE.  C[m,n] = sum_k x[m,k] * W[n,k]   (NT GEMM)
// Block = 256 threads = 8 waves; wave computes 32 rows x 64 cols
// (2 A-frags x 4 B-frags -> 8 accumulators, 12 b128 loads per 8 WMMAs).
// blockIdx.z selects mode: 0=Q, 1=K, 2=V.
//   Q: rope -> bf16 Qb[B,H,S,dk]
//   K: rope -> fp32 d_out K slot [B,S,H,dk] and bf16 Kb[B,H,S,dk]
//   V:          fp32 d_out V slot          and bf16 Vt[B,H,dk,S] (transposed)
// ---------------------------------------------------------------------------
__global__ __launch_bounds__(256)
void gemm_qkv_rope(const unsigned short* __restrict__ xb,
                   const unsigned short* __restrict__ Wqb,
                   const unsigned short* __restrict__ Wkb,
                   const unsigned short* __restrict__ Wvb,
                   const float* __restrict__ cosTab,   // [S, dk/2]
                   const float* __restrict__ sinTab,   // [S, dk/2]
                   float* __restrict__ outK, float* __restrict__ outV,
                   unsigned short* __restrict__ Qb,
                   unsigned short* __restrict__ Kb,
                   unsigned short* __restrict__ Vt) {
  const int mode = blockIdx.z;
  const unsigned short* W = (mode == 0) ? Wqb : ((mode == 1) ? Wkb : Wvb);

  const int w     = threadIdx.x >> 5;
  const int lane  = threadIdx.x & 31;
  const int hi    = lane >> 4;
  const int nl    = lane & 15;
  const int mbase = blockIdx.x * 64 + (w >> 2) * 32;    // 32 rows per wave
  const int nbase = blockIdx.y * 256 + (w & 3) * 64;    // 64 cols per wave

  const v8f zero8 = {0.f, 0.f, 0.f, 0.f, 0.f, 0.f, 0.f, 0.f};
  v8f acc[2][4] = {{zero8, zero8, zero8, zero8}, {zero8, zero8, zero8, zero8}};

  for (int kk = 0; kk < D_; kk += 32) {
    v16bf a0 = load_fragA(xb, D_, mbase,      kk);
    v16bf a1 = load_fragA(xb, D_, mbase + 16, kk);
    if (kk + 64 < D_) {  // prefetch next-but-one A tile (global_prefetch_b8)
      __builtin_prefetch(xb + (size_t)(mbase + nl) * D_ + kk + 64, 0, 1);
    }
#pragma unroll
    for (int j = 0; j < 4; ++j) {
      v16bf b = load_fragB(W, D_, nbase + 16 * j, kk);
      acc[0][j] = wmma_bf16(a0, b, acc[0][j]);
      acc[1][j] = wmma_bf16(a1, b, acc[1][j]);
    }
  }

#pragma unroll
  for (int mi = 0; mi < 2; ++mi) {
    const int mrow  = mbase + 16 * mi;
    const int bidx  = mrow / S_;     // batch (block never straddles a batch)
    const int srow0 = mrow % S_;     // sequence row of accumulator element r=0, hi=0

#pragma unroll
    for (int j = 0; j < 4; ++j) {
      const int n    = nbase + 16 * j + nl;   // global column in D
      const int head = n >> 7;                // dk = 128
      const int d    = n & (DK - 1);

      if (mode <= 1) {
        // RoPE on interleaved pairs (2i, 2i+1): adjacent columns -> lanes l, l^1.
        const int fi = d >> 1;
#pragma unroll
        for (int r = 0; r < 8; ++r) {
          const int   srow = srow0 + r + 8 * hi;
          const float c    = cosTab[srow * (DK / 2) + fi];
          const float s    = sinTab[srow * (DK / 2) + fi];
          const float v    = acc[mi][j][r];
          const float pp   = __shfl_xor(v, 1, 32);
          acc[mi][j][r] = (d & 1) ? (pp * s + v * c) : (v * c - pp * s);
        }
      }

      if (mode == 0) {
#pragma unroll
        for (int r = 0; r < 8; ++r) {
          const int srow = srow0 + r + 8 * hi;
          Qb[((size_t)(bidx * H_ + head) * S_ + srow) * DK + d] = f2bf(acc[mi][j][r]);
        }
      } else if (mode == 1) {
#pragma unroll
        for (int r = 0; r < 8; ++r) {
          const int srow = srow0 + r + 8 * hi;
          outK[((size_t)bidx * S_ + srow) * D_ + n] = acc[mi][j][r];
          Kb[((size_t)(bidx * H_ + head) * S_ + srow) * DK + d] = f2bf(acc[mi][j][r]);
        }
      } else {
        unsigned short tmp[8];
#pragma unroll
        for (int r = 0; r < 8; ++r) {
          const int srow = srow0 + r + 8 * hi;
          outV[((size_t)bidx * S_ + srow) * D_ + n] = acc[mi][j][r];
          tmp[r] = f2bf(acc[mi][j][r]);
        }
        // Vt row (fixed d) gets 8 consecutive sequence positions -> one b128 store.
        *(uint4*)(Vt + ((size_t)(bidx * H_ + head) * DK + d) * S_ + srow0 + 8 * hi) =
            *(const uint4*)tmp;
      }
    }
  }
}

// ---------------------------------------------------------------------------
// Flash-style causal attention. Grid = (S/128, B*H); block = 8 waves.
// Each wave owns 16 queries. Scores are computed TRANSPOSED (A = K-tile,
// B = Q) so key-dim reductions are in-lane, and the exp'd score C-fragments
// map exactly onto the A-fragment layout of the P*V WMMA (no lane shuffles).
// ---------------------------------------------------------------------------
__global__ __launch_bounds__(256)
void attn_kernel(const unsigned short* __restrict__ Qb,   // [B,H,S,dk]
                 const unsigned short* __restrict__ Kb,   // [B,H,S,dk]
                 const unsigned short* __restrict__ Vt,   // [B,H,dk,S]
                 unsigned short* __restrict__ ctxb) {     // [B,S,D]
  const int bh    = blockIdx.y;
  const int w     = threadIdx.x >> 5;
  const int lane  = threadIdx.x & 31;
  const int hi    = lane >> 4;
  const int nl    = lane & 15;
  const int qbase = blockIdx.x * 128 + w * 16;

  const unsigned short* Qh = Qb + (size_t)bh * S_ * DK;
  const unsigned short* Kh = Kb + (size_t)bh * S_ * DK;
  const unsigned short* Vh = Vt + (size_t)bh * DK * S_;

  // Q fragments: B-operand of the transposed-score WMMA (queries as columns).
  v16bf qf[4];
#pragma unroll
  for (int c = 0; c < 4; ++c) qf[c] = load_fragB(Qh, DK, qbase, 32 * c);

  const v8f zero8 = {0.f, 0.f, 0.f, 0.f, 0.f, 0.f, 0.f, 0.f};
  v8f ctx[8] = {zero8, zero8, zero8, zero8, zero8, zero8, zero8, zero8};

  float m    = -1e30f;   // running max for query q = lane&15 (dup. in lane pairs)
  float lsum = 0.f;
  const float scale = 0.08838834764831845f;  // 1/sqrt(128)
  const int   qg    = qbase + nl;

  for (int kb = 0; kb <= qbase + 15; kb += 32) {
    v8f sc0 = zero8, sc1 = zero8;   // scores^T for keys kb..kb+15 / kb+16..kb+31
#pragma unroll
    for (int c = 0; c < 4; ++c) {
      v16bf k0 = load_fragA(Kh, DK, kb,      32 * c);
      v16bf k1 = load_fragA(Kh, DK, kb + 16, 32 * c);
      sc0 = wmma_bf16(k0, qf[c], sc0);
      sc1 = wmma_bf16(k1, qf[c], sc1);
    }

    const bool need_mask = (kb + 31 > qbase);
    float bm = -1e30f;
#pragma unroll
    for (int r = 0; r < 8; ++r) {
      float v0 = sc0[r] * scale;
      float v1 = sc1[r] * scale;
      if (need_mask) {
        const int key0 = kb + r + 8 * hi;   // key row of this C element
        if (key0 > qg)      v0 = -1e30f;
        if (key0 + 16 > qg) v1 = -1e30f;
      }
      sc0[r] = v0; sc1[r] = v1;
      bm = fmaxf(bm, fmaxf(v0, v1));
    }
    bm = fmaxf(bm, __shfl_xor(bm, 16, 32));   // combine the two key half-tiles
    const float mnew  = fmaxf(m, bm);
    const float alpha = __expf(m - mnew);
    m = mnew;

    BFrag16 pf;
    float rs = 0.f;
#pragma unroll
    for (int r = 0; r < 8; ++r) {
      const float p0 = __expf(sc0[r] - mnew);
      const float p1 = __expf(sc1[r] - mnew);
      rs += p0 + p1;
      pf.h[r]     = f2bf(p0);   // A-frag elems 0..7  <- keys kb +  r + 8*hi
      pf.h[8 + r] = f2bf(p1);   // A-frag elems 8..15 <- keys kb+16+r + 8*hi
    }
    rs += __shfl_xor(rs, 16, 32);
    lsum = lsum * alpha + rs;

    // Rescale ctx accumulators: factor is per query-row = accumulator element.
#pragma unroll
    for (int r = 0; r < 8; ++r) {
      const float ar = __shfl(alpha, r + 8 * hi, 32);
#pragma unroll
      for (int n = 0; n < 8; ++n) ctx[n][r] *= ar;
    }

    // ctx[q, d] += P[q, key] * V[key, d]  — 8 d-tiles of 16.
#pragma unroll
    for (int n = 0; n < 8; ++n) {
      v16bf vf = load_fragB(Vh, S_, 16 * n, kb);   // contiguous keys from Vt
      ctx[n] = wmma_bf16(pf.v, vf, ctx[n]);
    }
  }

  // Normalize and store bf16 context in [B,S,D] layout for the output GEMM.
  const int bidx = bh / H_;
  const int hh   = bh % H_;
#pragma unroll
  for (int r = 0; r < 8; ++r) {
    const float lr  = __shfl(lsum, r + 8 * hi, 32);
    const float inv = 1.0f / lr;
    const int   q   = qbase + r + 8 * hi;
#pragma unroll
    for (int n = 0; n < 8; ++n) {
      ctxb[((size_t)bidx * S_ + q) * D_ + hh * DK + 16 * n + nl] =
          f2bf(ctx[n][r] * inv);
    }
  }
}

// ---------------------------------------------------------------------------
// Output projection: out[m,n] = sum_k ctx[m,k] * Wo[n,k]  (fp32 result)
// Wave tile 32x64 like the QKV GEMM.
// ---------------------------------------------------------------------------
__global__ __launch_bounds__(256)
void gemm_out_kernel(const unsigned short* __restrict__ ctxb,
                     const unsigned short* __restrict__ Wob,
                     float* __restrict__ out) {
  const int w     = threadIdx.x >> 5;
  const int lane  = threadIdx.x & 31;
  const int hi    = lane >> 4;
  const int nl    = lane & 15;
  const int mbase = blockIdx.x * 64 + (w >> 2) * 32;
  const int nbase = blockIdx.y * 256 + (w & 3) * 64;

  const v8f zero8 = {0.f, 0.f, 0.f, 0.f, 0.f, 0.f, 0.f, 0.f};
  v8f acc[2][4] = {{zero8, zero8, zero8, zero8}, {zero8, zero8, zero8, zero8}};

  for (int kk = 0; kk < D_; kk += 32) {
    v16bf a0 = load_fragA(ctxb, D_, mbase,      kk);
    v16bf a1 = load_fragA(ctxb, D_, mbase + 16, kk);
    if (kk + 64 < D_) {
      __builtin_prefetch(ctxb + (size_t)(mbase + nl) * D_ + kk + 64, 0, 1);
    }
#pragma unroll
    for (int j = 0; j < 4; ++j) {
      v16bf b = load_fragB(Wob, D_, nbase + 16 * j, kk);
      acc[0][j] = wmma_bf16(a0, b, acc[0][j]);
      acc[1][j] = wmma_bf16(a1, b, acc[1][j]);
    }
  }

#pragma unroll
  for (int mi = 0; mi < 2; ++mi) {
#pragma unroll
    for (int j = 0; j < 4; ++j) {
      const int n = nbase + 16 * j + nl;
#pragma unroll
      for (int r = 0; r < 8; ++r) {
        out[((size_t)mbase + 16 * mi + r + 8 * hi) * D_ + n] = acc[mi][j][r];
      }
    }
  }
}

// ---------------------------------------------------------------------------
extern "C" void kernel_launch(void* const* d_in, const int* in_sizes, int n_in,
                              void* d_out, int out_size, void* d_ws, size_t ws_size,
                              hipStream_t stream) {
  const float* x   = (const float*)d_in[0];
  const float* fc  = (const float*)d_in[1];   // freqs_cos [S, dk/2]
  const float* fs  = (const float*)d_in[2];   // freqs_sin [S, dk/2]
  // d_in[3] = additive causal mask — handled analytically, unused.
  const float* Wq  = (const float*)d_in[4];
  const float* Wk  = (const float*)d_in[5];
  const float* Wv  = (const float*)d_in[6];
  const float* Wo  = (const float*)d_in[7];

  float* out  = (float*)d_out;                      // output [B,S,D]
  float* outK = out  + (size_t)B_ * S_ * D_;        // K cache [B,S,H,dk] == [B,S,D]
  float* outV = outK + (size_t)B_ * S_ * D_;        // V cache

  // Workspace carve-up (~117 MB)
  char* p = (char*)d_ws;
  auto take = [&](size_t bytes) {
    char* r = p;
    p += (bytes + 255) & ~(size_t)255;
    return r;
  };
  unsigned short* xb   = (unsigned short*)take((size_t)M_ * D_ * 2);
  unsigned short* Wqb  = (unsigned short*)take((size_t)D_ * D_ * 2);
  unsigned short* Wkb  = (unsigned short*)take((size_t)D_ * D_ * 2);
  unsigned short* Wvb  = (unsigned short*)take((size_t)D_ * D_ * 2);
  unsigned short* Wob  = (unsigned short*)take((size_t)D_ * D_ * 2);
  unsigned short* Qb   = (unsigned short*)take((size_t)M_ * D_ * 2);
  unsigned short* Kb   = (unsigned short*)take((size_t)M_ * D_ * 2);
  unsigned short* Vt   = (unsigned short*)take((size_t)M_ * D_ * 2);
  unsigned short* ctxb = (unsigned short*)take((size_t)M_ * D_ * 2);

  // 1) fp32 -> bf16 conversions
  {
    long n4x = (long)M_ * D_ / 4;   // 2,097,152
    long n4w = (long)D_ * D_ / 4;   // 1,048,576
    cvt_f32_bf16<<<(unsigned)(n4x / 256), 256, 0, stream>>>(x,  xb,  n4x);
    cvt_f32_bf16<<<(unsigned)(n4w / 256), 256, 0, stream>>>(Wq, Wqb, n4w);
    cvt_f32_bf16<<<(unsigned)(n4w / 256), 256, 0, stream>>>(Wk, Wkb, n4w);
    cvt_f32_bf16<<<(unsigned)(n4w / 256), 256, 0, stream>>>(Wv, Wvb, n4w);
    cvt_f32_bf16<<<(unsigned)(n4w / 256), 256, 0, stream>>>(Wo, Wob, n4w);
  }

  // 2) QKV projections + RoPE (z-dim selects Q/K/V)
  gemm_qkv_rope<<<dim3(M_ / 64, D_ / 256, 3), 256, 0, stream>>>(
      xb, Wqb, Wkb, Wvb, fc, fs, outK, outV, Qb, Kb, Vt);

  // 3) causal flash attention
  attn_kernel<<<dim3(S_ / 128, B_ * H_), 256, 0, stream>>>(Qb, Kb, Vt, ctxb);

  // 4) output projection
  gemm_out_kernel<<<dim3(M_ / 64, D_ / 256), 256, 0, stream>>>(ctxb, Wob, out);
}
